// fancy_GNN_45002667327985
// MI455X (gfx1250) — compile-verified, matrix operand order checked by reading, not run on hardware
//
#include <hip/hip_runtime.h>
#include <hip/hip_bf16.h>

// ---------------------------------------------------------------------------
// Problem constants (match reference)
// ---------------------------------------------------------------------------
#define HCH   128
#define NLAY  2
#define NEDGE 400000
#define NATOM 600000
#define NCHEM 20000
#define NGENE 20000
#define NASSY 10000
#define NREL  13
#define OUTC  2
#define NSLOPE 0.2f

typedef __attribute__((ext_vector_type(16))) __bf16 v16bf;
typedef __attribute__((ext_vector_type(8)))  __bf16 v8bf;
typedef __attribute__((ext_vector_type(4)))  __bf16 v4bf;
typedef __attribute__((ext_vector_type(8)))  float  v8f;

__device__ inline void atomicMaxF(float* addr, float v) {
    if (v >= 0.0f) atomicMax((int*)addr, __float_as_int(v));
    else           atomicMin((unsigned int*)addr, __float_as_uint(v));
}

// ---------------------------------------------------------------------------
// Generic fill
// ---------------------------------------------------------------------------
__global__ void fill_f32(float* __restrict__ p, float v, int n) {
    int i = blockIdx.x * blockDim.x + threadIdx.x;
    if (i < n) p[i] = v;
}

// f32 -> bf16 elementwise (n multiple of 4); float4 in, 4x bf16 (8B) out
__global__ void cvt_bf16(const float* __restrict__ src, __bf16* __restrict__ dst, int n4) {
    int i = blockIdx.x * blockDim.x + threadIdx.x;
    if (i >= n4) return;
    float4 v = ((const float4*)src)[i];
    v4bf o;
    o[0] = (__bf16)v.x; o[1] = (__bf16)v.y; o[2] = (__bf16)v.z; o[3] = (__bf16)v.w;
    ((v4bf*)dst)[i] = o;
}

// W[128x128] f32 row-major -> Wt[128x128] bf16 with Wt[n*128+k] = W[k*128+n]
__global__ void wt_cvt(const float* __restrict__ W, __bf16* __restrict__ Wt) {
    int i = blockIdx.x * blockDim.x + threadIdx.x;
    if (i >= HCH * HCH) return;
    int n = i >> 7, k = i & (HCH - 1);
    Wt[i] = (__bf16)W[k * HCH + n];
}

// ---------------------------------------------------------------------------
// GEMM: Y = A @ W + bias, A given as bf16 row-major [nrows x 128],
// W given transposed bf16 [128 x 128] (Wt[n][k]). One wave computes a
// 16-row x 128-col stripe: 8 f32 accumulator tiles, 4 K-chunks of
// v_wmma_f32_16x16x32_bf16. Every fragment half-slice = one b128 load.
// ---------------------------------------------------------------------------
__global__ __launch_bounds__(32)
void gemm128_bias(const __bf16* __restrict__ Abf, const __bf16* __restrict__ Wt,
                  const float* __restrict__ bias, float* __restrict__ Y, int nrows)
{
    const int lane = threadIdx.x & 31;
    const int r    = lane & 15;       // A-row / B-col / D-col for this lane
    const int half = lane >> 4;
    const int row0 = blockIdx.x * 16;
    if (row0 >= nrows) return;

    v8f acc[8] = {};
    const v8bf* ap = (const v8bf*)(Abf + (size_t)(row0 + r) * HCH);

    #pragma unroll
    for (int kc = 0; kc < 4; ++kc) {
        const int kb = kc * 32 + half * 8;        // multiple of 8
        const int ki = kb >> 3;
        const v8bf a0 = ap[ki];
        const v8bf a1 = ap[ki + 2];               // K = kb+16 .. kb+23
        const v16bf a = __builtin_shufflevector(a0, a1,
            0, 1, 2, 3, 4, 5, 6, 7, 8, 9, 10, 11, 12, 13, 14, 15);

        #pragma unroll
        for (int ct = 0; ct < 8; ++ct) {
            const v8bf* bp = (const v8bf*)(Wt + (size_t)(ct * 16 + r) * HCH);
            const v8bf b0 = bp[ki];
            const v8bf b1 = bp[ki + 2];
            const v16bf b = __builtin_shufflevector(b0, b1,
                0, 1, 2, 3, 4, 5, 6, 7, 8, 9, 10, 11, 12, 13, 14, 15);
            acc[ct] = __builtin_amdgcn_wmma_f32_16x16x32_bf16(
                false, a, false, b, (short)0, acc[ct], false, false);
        }
    }

    #pragma unroll
    for (int ct = 0; ct < 8; ++ct) {
        const int n  = ct * 16 + r;
        const float bv = bias[n];
        #pragma unroll
        for (int i = 0; i < 8; ++i) {
            const int m = i + 8 * half;
            Y[(size_t)(row0 + m) * HCH + n] = acc[ct][i] + bv;
        }
    }
}

// ---------------------------------------------------------------------------
// Fused atom encoder: agg[hyper_dst[r], :] += (x_atom @ W_mol)[r, :]
// x_atom streamed once (f32 float4 loads, native cvt to bf16); W_mol
// pre-transposed bf16. Scatter via f32 global atomics.
// ---------------------------------------------------------------------------
__global__ __launch_bounds__(32)
void atom_gemm_scatter(const float* __restrict__ A, const __bf16* __restrict__ Wt,
                       const int* __restrict__ hyper_dst, float* __restrict__ agg,
                       int nrows)
{
    const int lane = threadIdx.x & 31;
    const int r    = lane & 15;
    const int half = lane >> 4;
    const int row0 = blockIdx.x * 16;
    if (row0 >= nrows) return;

    v8f acc[8] = {};
    const float4* ap = (const float4*)(A + (size_t)(row0 + r) * HCH);

    #pragma unroll
    for (int kc = 0; kc < 4; ++kc) {
        const int kb = kc * 32 + half * 8;
        const int fi = kb >> 2;
        const float4 f0 = ap[fi];
        const float4 f1 = ap[fi + 1];
        const float4 f2 = ap[fi + 4];             // K = kb+16
        const float4 f3 = ap[fi + 5];
        v16bf a;
        a[0]  = (__bf16)f0.x; a[1]  = (__bf16)f0.y; a[2]  = (__bf16)f0.z; a[3]  = (__bf16)f0.w;
        a[4]  = (__bf16)f1.x; a[5]  = (__bf16)f1.y; a[6]  = (__bf16)f1.z; a[7]  = (__bf16)f1.w;
        a[8]  = (__bf16)f2.x; a[9]  = (__bf16)f2.y; a[10] = (__bf16)f2.z; a[11] = (__bf16)f2.w;
        a[12] = (__bf16)f3.x; a[13] = (__bf16)f3.y; a[14] = (__bf16)f3.z; a[15] = (__bf16)f3.w;

        const int ki = kb >> 3;
        #pragma unroll
        for (int ct = 0; ct < 8; ++ct) {
            const v8bf* bp = (const v8bf*)(Wt + (size_t)(ct * 16 + r) * HCH);
            const v8bf b0 = bp[ki];
            const v8bf b1 = bp[ki + 2];
            const v16bf b = __builtin_shufflevector(b0, b1,
                0, 1, 2, 3, 4, 5, 6, 7, 8, 9, 10, 11, 12, 13, 14, 15);
            acc[ct] = __builtin_amdgcn_wmma_f32_16x16x32_bf16(
                false, a, false, b, (short)0, acc[ct], false, false);
        }
    }

    int chem[8];
    #pragma unroll
    for (int i = 0; i < 8; ++i) chem[i] = hyper_dst[row0 + i + 8 * half];

    #pragma unroll
    for (int ct = 0; ct < 8; ++ct) {
        const int n = ct * 16 + r;
        #pragma unroll
        for (int i = 0; i < 8; ++i)
            atomicAdd(&agg[(size_t)chem[i] * HCH + n], acc[ct][i]);
    }
}

__global__ void count_kernel(const int* __restrict__ hyper_dst, float* __restrict__ cnt, int n) {
    int i = blockIdx.x * blockDim.x + threadIdx.x;
    if (i < n) atomicAdd(&cnt[hyper_dst[i]], 1.0f);
}

__global__ void combine_chem(const float* __restrict__ agg, const float* __restrict__ cnt,
                             const float* __restrict__ x_chem, float* __restrict__ xs0, int n) {
    int i = blockIdx.x * blockDim.x + threadIdx.x;
    if (i < n) xs0[i] = agg[i] / fmaxf(cnt[i >> 7], 1.0f) + x_chem[i];
}

// ---------------------------------------------------------------------------
// GATv2 edge pass 1: e = leaky_relu(xl[src]+xr[dst]) . a ; segment max to m
// ---------------------------------------------------------------------------
__global__ __launch_bounds__(256)
void edge_logits(const float* __restrict__ xl, const float* __restrict__ xr,
                 const int* __restrict__ src, const int* __restrict__ dst,
                 const float* __restrict__ av, float* __restrict__ e,
                 float* __restrict__ m, int nedges)
{
    const int w    = blockIdx.x * 8 + (threadIdx.x >> 5);
    const int lane = threadIdx.x & 31;
    if (w >= nedges) return;
    const int s = src[w], d = dst[w];
    const float4 vl = ((const float4*)(xl + (size_t)s * HCH))[lane];
    const float4 vr = ((const float4*)(xr + (size_t)d * HCH))[lane];
    const float4 va = ((const float4*)av)[lane];
    float t, sum = 0.0f;
    t = vl.x + vr.x; sum += (t > 0.0f ? t : NSLOPE * t) * va.x;
    t = vl.y + vr.y; sum += (t > 0.0f ? t : NSLOPE * t) * va.y;
    t = vl.z + vr.z; sum += (t > 0.0f ? t : NSLOPE * t) * va.z;
    t = vl.w + vr.w; sum += (t > 0.0f ? t : NSLOPE * t) * va.w;
    #pragma unroll
    for (int off = 16; off > 0; off >>= 1) sum += __shfl_xor(sum, off, 32);
    if (lane == 0) { e[w] = sum; atomicMaxF(&m[d], sum); }
}

// pass 2: ex = exp(e - m[dst]); denom[dst] += ex   (in-place e <- ex)
__global__ void edge_expsum(float* __restrict__ e, const int* __restrict__ dst,
                            const float* __restrict__ m, float* __restrict__ denom, int nedges)
{
    int i = blockIdx.x * blockDim.x + threadIdx.x;
    if (i >= nedges) return;
    const int d = dst[i];
    const float ex = __expf(e[i] - m[d]);
    e[i] = ex;
    atomicAdd(&denom[d], ex);
}

// pass 3: agg[dst,:] += (ex/denom[dst]) * xl[src,:]
__global__ __launch_bounds__(256)
void edge_scatter(const float* __restrict__ xl, const float* __restrict__ ex,
                  const float* __restrict__ denom, const int* __restrict__ src,
                  const int* __restrict__ dst, float* __restrict__ agg, int nedges)
{
    const int w    = blockIdx.x * 8 + (threadIdx.x >> 5);
    const int lane = threadIdx.x & 31;
    if (w >= nedges) return;
    const int s = src[w], d = dst[w];
    const float alpha = ex[w] / fmaxf(denom[d], 1e-16f);
    const float4 vl = ((const float4*)(xl + (size_t)s * HCH))[lane];
    float* base = agg + (size_t)d * HCH + lane * 4;
    atomicAdd(base + 0, alpha * vl.x);
    atomicAdd(base + 1, alpha * vl.y);
    atomicAdd(base + 2, alpha * vl.z);
    atomicAdd(base + 3, alpha * vl.w);
}

// per-relation output bias cb broadcast-added to every dst row
__global__ void add_row_bias(float* __restrict__ agg, const float* __restrict__ cb, int n) {
    int i = blockIdx.x * blockDim.x + threadIdx.x;
    if (i < n) agg[i] += cb[i & (HCH - 1)];
}

// ---------------------------------------------------------------------------
// LayerNorm + ReLU, one wave per row
// ---------------------------------------------------------------------------
__global__ __launch_bounds__(256)
void ln_relu(const float* __restrict__ agg, const float* __restrict__ g,
             const float* __restrict__ b, float* __restrict__ out, int nrows)
{
    const int row  = blockIdx.x * 8 + (threadIdx.x >> 5);
    const int lane = threadIdx.x & 31;
    if (row >= nrows) return;
    const float4 v = ((const float4*)(agg + (size_t)row * HCH))[lane];
    float s  = v.x + v.y + v.z + v.w;
    float ss = v.x * v.x + v.y * v.y + v.z * v.z + v.w * v.w;
    #pragma unroll
    for (int off = 16; off > 0; off >>= 1) {
        s  += __shfl_xor(s,  off, 32);
        ss += __shfl_xor(ss, off, 32);
    }
    const float mu  = s * (1.0f / HCH);
    const float var = ss * (1.0f / HCH) - mu * mu;
    const float inv = rsqrtf(var + 1e-5f);
    const float4 gg = ((const float4*)g)[lane];
    const float4 bb = ((const float4*)b)[lane];
    float4 o;
    o.x = fmaxf((v.x - mu) * inv * gg.x + bb.x, 0.0f);
    o.y = fmaxf((v.y - mu) * inv * gg.y + bb.y, 0.0f);
    o.z = fmaxf((v.z - mu) * inv * gg.z + bb.z, 0.0f);
    o.w = fmaxf((v.w - mu) * inv * gg.w + bb.w, 0.0f);
    ((float4*)(out + (size_t)row * HCH))[lane] = o;
}

// ---------------------------------------------------------------------------
// Final prediction head: out = xs0 @ Wp + bp   (OUT=2, thread per row)
// ---------------------------------------------------------------------------
__global__ void pred_head(const float* __restrict__ xs0, const float* __restrict__ Wp,
                          const float* __restrict__ bp, float* __restrict__ out, int nrows)
{
    int i = blockIdx.x * blockDim.x + threadIdx.x;
    if (i >= nrows) return;
    float a0 = bp[0], a1 = bp[1];
    const float* xr = xs0 + (size_t)i * HCH;
    #pragma unroll 4
    for (int k = 0; k < HCH; ++k) {
        const float x = xr[k];
        a0 += x * Wp[k * OUTC + 0];
        a1 += x * Wp[k * OUTC + 1];
    }
    out[i * OUTC + 0] = a0;
    out[i * OUTC + 1] = a1;
}

// ---------------------------------------------------------------------------
// Host orchestration
// ---------------------------------------------------------------------------
extern "C" void kernel_launch(void* const* d_in, const int* in_sizes, int n_in,
                              void* d_out, int out_size, void* d_ws, size_t ws_size,
                              hipStream_t stream)
{
    const float* x_atom  = (const float*)d_in[0];
    const float* x_chem  = (const float*)d_in[1];
    const float* x_gene  = (const float*)d_in[2];
    const float* x_assay = (const float*)d_in[3];
    const float* W_mol   = (const float*)d_in[4];
    const float* Wl      = (const float*)d_in[5];
    const float* Wr      = (const float*)d_in[6];
    const float* bl      = (const float*)d_in[7];
    const float* br      = (const float*)d_in[8];
    const float* att_a   = (const float*)d_in[9];
    const float* cb      = (const float*)d_in[10];
    const float* ln_g    = (const float*)d_in[11];
    const float* ln_b    = (const float*)d_in[12];
    const float* Wp      = (const float*)d_in[13];
    const float* bp      = (const float*)d_in[14];
    const int* hyper_dst = (const int*)d_in[15];
    const int* eidx      = (const int*)d_in[16];
    float* out           = (float*)d_out;

    static const int rel_src[NREL] = {0,0,0,0,0,2,1,2,2,1,1,1,1};
    static const int rel_dst[NREL] = {2,2,1,1,1,1,1,0,0,0,0,0,2};
    const int Nt[3] = {NCHEM, NGENE, NASSY};

    // workspace layout (float units)
    float* ws = (float*)d_ws;
    size_t o = 0;
    float* cnt   = ws + o; o += NCHEM;
    float* mbuf  = ws + o; o += NCHEM;                 // >= max dst count
    float* denom = ws + o; o += NCHEM;
    float* xs0   = ws + o; o += (size_t)NCHEM * HCH;
    float* xs1   = ws + o; o += (size_t)NGENE * HCH;
    float* xs2   = ws + o; o += (size_t)NASSY * HCH;
    float* agg0  = ws + o; o += (size_t)NCHEM * HCH;
    float* agg1  = ws + o; o += (size_t)NGENE * HCH;
    float* agg2  = ws + o; o += (size_t)NASSY * HCH;
    float* xlbuf = ws + o; o += (size_t)NCHEM * HCH;   // >= max src count
    float* xrbuf = ws + o; o += (size_t)NCHEM * HCH;   // >= max dst count
    float* ebuf  = ws + o; o += (size_t)NEDGE;
    __bf16* xsbf0 = (__bf16*)(ws + o); o += (size_t)NCHEM * HCH / 2;
    __bf16* xsbf1 = (__bf16*)(ws + o); o += (size_t)NGENE * HCH / 2;
    __bf16* xsbf2 = (__bf16*)(ws + o); o += (size_t)NASSY * HCH / 2;
    __bf16* wmolT = (__bf16*)(ws + o); o += HCH * HCH / 2;
    __bf16* wtl   = (__bf16*)(ws + o); o += HCH * HCH / 2;
    __bf16* wtr   = (__bf16*)(ws + o); o += HCH * HCH / 2;

    const float NEGINF = -3.402823466e38f;

    // ---- bottom_up: atom encoder + mean aggregate into Chemical -----------
    wt_cvt<<<(HCH * HCH + 255) / 256, 256, 0, stream>>>(W_mol, wmolT);
    fill_f32<<<(NCHEM + 255) / 256, 256, 0, stream>>>(cnt, 0.0f, NCHEM);
    fill_f32<<<(NCHEM * HCH + 255) / 256, 256, 0, stream>>>(agg0, 0.0f, NCHEM * HCH);
    atom_gemm_scatter<<<NATOM / 16, 32, 0, stream>>>(x_atom, wmolT, hyper_dst, agg0, NATOM);
    count_kernel<<<(NATOM + 255) / 256, 256, 0, stream>>>(hyper_dst, cnt, NATOM);
    combine_chem<<<(NCHEM * HCH + 255) / 256, 256, 0, stream>>>(agg0, cnt, x_chem, xs0, NCHEM * HCH);

    const float* xs_in[3]  = {xs0, x_gene, x_assay};
    float*       xs_out[3] = {xs0, xs1, xs2};
    __bf16*      xsbf[3]   = {xsbf0, xsbf1, xsbf2};
    float*       agg[3]    = {agg0, agg1, agg2};

    for (int l = 0; l < NLAY; ++l) {
        for (int t = 0; t < 3; ++t) {
            cvt_bf16<<<(Nt[t] * HCH / 4 + 255) / 256, 256, 0, stream>>>(
                xs_in[t], xsbf[t], Nt[t] * HCH / 4);
            fill_f32<<<(Nt[t] * HCH + 255) / 256, 256, 0, stream>>>(agg[t], 0.0f, Nt[t] * HCH);
        }

        for (int r = 0; r < NREL; ++r) {
            const int s = rel_src[r], d = rel_dst[r];
            const size_t wo = ((size_t)l * NREL + r) * HCH * HCH;
            const size_t vo = ((size_t)l * NREL + r) * HCH;
            const int* srcp = eidx + ((size_t)r * 2 + 0) * NEDGE;
            const int* dstp = eidx + ((size_t)r * 2 + 1) * NEDGE;

            wt_cvt<<<(HCH * HCH + 255) / 256, 256, 0, stream>>>(Wl + wo, wtl);
            wt_cvt<<<(HCH * HCH + 255) / 256, 256, 0, stream>>>(Wr + wo, wtr);
            gemm128_bias<<<Nt[s] / 16, 32, 0, stream>>>(xsbf[s], wtl, bl + vo, xlbuf, Nt[s]);
            gemm128_bias<<<Nt[d] / 16, 32, 0, stream>>>(xsbf[d], wtr, br + vo, xrbuf, Nt[d]);

            fill_f32<<<(Nt[d] + 255) / 256, 256, 0, stream>>>(mbuf, NEGINF, Nt[d]);
            fill_f32<<<(Nt[d] + 255) / 256, 256, 0, stream>>>(denom, 0.0f, Nt[d]);

            edge_logits<<<(NEDGE + 7) / 8, 256, 0, stream>>>(xlbuf, xrbuf, srcp, dstp,
                                                             att_a + vo, ebuf, mbuf, NEDGE);
            edge_expsum<<<(NEDGE + 255) / 256, 256, 0, stream>>>(ebuf, dstp, mbuf, denom, NEDGE);
            edge_scatter<<<(NEDGE + 7) / 8, 256, 0, stream>>>(xlbuf, ebuf, denom, srcp, dstp,
                                                              agg[d], NEDGE);
            add_row_bias<<<(Nt[d] * HCH + 255) / 256, 256, 0, stream>>>(agg[d], cb + vo, Nt[d] * HCH);
        }

        for (int t = 0; t < 3; ++t) {
            const size_t go = ((size_t)l * 3 + t) * HCH;
            ln_relu<<<(Nt[t] + 7) / 8, 256, 0, stream>>>(agg[t], ln_g + go, ln_b + go,
                                                         xs_out[t], Nt[t]);
        }
        xs_in[0] = xs0; xs_in[1] = xs1; xs_in[2] = xs2;
    }

    pred_head<<<(NCHEM + 255) / 256, 256, 0, stream>>>(xs0, Wp, bp, out, NCHEM);
}
